// VelocityWarpper_87909390614848
// MI455X (gfx1250) — matrix operand order repaired and checked by reading, not compiled, not run on hardware
//
#include <hip/hip_runtime.h>
#include <hip/hip_bf16.h>
#include <math.h>

typedef __attribute__((ext_vector_type(16))) _Float16 v16h;
typedef __attribute__((ext_vector_type(8)))  _Float16 v8h;
typedef __attribute__((ext_vector_type(8)))  float    v8f;

#define NPTS      131072
#define CODE_DIM  64
#define HID       128
#define IN_DIM    68
#define DT_MAX    0.125f
#define N_STEPS   8
#define WAVES     4
#define PTS_BLK   (WAVES * 16)

// f16 transposed weight layout (units: halves) -- same in d_ws and in LDS
#define W1T_OFF 0                      // [n=128][k=96], k>=68 zero-padded
#define W2T_OFF (128 * 96)             // [n=128][k=128]
#define W3T_OFF (W2T_OFF + 128 * 128)  // [n=16][k=128], n>=3 zero-padded
#define W_HALves (W3T_OFF + 16 * 128)  // 30720 halves = 61440 bytes
#define W_BYTES  (W_HALves * 2)

// dynamic LDS carve
#define SACT_BYTES (WAVES * 16 * HID * 2)  // 16384
#define SC_BYTES   (WAVES * 16 * 16 * 4)   // 4096
#define SMEM_BYTES (W_BYTES + SACT_BYTES + SC_BYTES)  // 81920 (<< 320KB/WGP)

#if __has_builtin(__builtin_amdgcn_tanhf)
#define TANHF(x) __builtin_amdgcn_tanhf(x)
#else
#define TANHF(x) tanhf(x)
#endif

// ---------------- weight prep: f32 row-major -> f16 transposed ----------------
__global__ void prep_weights(const float* __restrict__ W1,
                             const float* __restrict__ W2,
                             const float* __restrict__ W3,
                             _Float16* __restrict__ ws) {
  int stride = gridDim.x * blockDim.x;
  int i0 = blockIdx.x * blockDim.x + threadIdx.x;
  for (int i = i0; i < 128 * 96; i += stride) {
    int n = i / 96, k = i % 96;
    ws[W1T_OFF + i] = (_Float16)((k < IN_DIM) ? W1[k * HID + n] : 0.0f);
  }
  for (int i = i0; i < 128 * 128; i += stride) {
    int n = i >> 7, k = i & 127;
    ws[W2T_OFF + i] = (_Float16)W2[k * HID + n];
  }
  for (int i = i0; i < 16 * 128; i += stride) {
    int n = i >> 7, k = i & 127;
    ws[W3T_OFF + i] = (_Float16)((n < 3) ? W3[k * 3 + n] : 0.0f);
  }
}

// ---------------- fragment helpers ----------------
__device__ __attribute__((always_inline)) v16h ld16(const _Float16* p0,
                                                    const _Float16* p1) {
  v8h lo = *(const v8h*)p0;
  v8h hi = *(const v8h*)p1;
  v16h r;
#pragma unroll
  for (int i = 0; i < 8; ++i) { r[i] = lo[i]; r[i + 8] = hi[i]; }
  return r;
}

// A fragment (16x32 f16): lane l<16 holds row l, K in {c..c+7, c+16..c+23};
// lane>=16 holds row l-16, K in {c+8..c+15, c+24..c+31}.
__device__ __attribute__((always_inline)) v16h afrag(const _Float16* actrow,
                                                     int kc, int hl) {
  const _Float16* s = actrow + kc * 32 + hl * 8;
  return ld16(s, s + 16);
}

// B fragment (32x16 f16): lane holds column n = l&15, K in {c+8h..c+8h+7,
// c+16+8h..c+16+8h+7} (h = l>>4); weight stored transposed [n][k].
__device__ __attribute__((always_inline)) v16h bfrag(const _Float16* wT,
                                                     int stride, int n, int kc,
                                                     int hl) {
  const _Float16* s = wT + n * stride + kc * 32 + hl * 8;
  return ld16(s, s + 16);
}

__device__ __attribute__((always_inline)) v8f wmma_f16(v16h a, v16h b, v8f c) {
  return __builtin_amdgcn_wmma_f32_16x16x32_f16(false, a, false, b, (short)0, c,
                                                false, false);
}

__device__ __attribute__((always_inline)) v8f tanh8(v8f x) {
  v8f r;
#pragma unroll
  for (int i = 0; i < 8; ++i) r[i] = TANHF(x[i]);
  return r;
}

// C layout: vgpr r, lane l -> row M = r + 8*(l>>4), col N = l&15
__device__ __attribute__((always_inline)) void store_act(_Float16* act, v8f c,
                                                         int nt, int l) {
  int n = nt * 16 + (l & 15);
  int mb = (l >> 4) * 8;
#pragma unroll
  for (int r = 0; r < 8; ++r) act[(mb + r) * HID + n] = (_Float16)c[r];
}

__device__ __attribute__((always_inline)) void store_c(float* cc, v8f c, int l) {
  int n = l & 15;
  int mb = (l >> 4) * 8;
#pragma unroll
  for (int r = 0; r < 8; ++r) cc[(mb + r) * 16 + n] = c[r];
}

__device__ __attribute__((always_inline)) void ds_fence() {
  asm volatile("s_wait_dscnt 0" ::: "memory");
}

// ---------------- full MLP forward for a 16-point wave tile ----------------
__device__ __attribute__((always_inline)) void mlp_eval(
    const _Float16* sW, const float* __restrict__ b1,
    const float* __restrict__ b2, const float* __restrict__ b3, _Float16* act,
    float* cc, int l, v16h a0, v16h a1, v16h a2, const v16h* w3f, v16h* h1f,
    v8f* h2t, float* vel) {
  int hl = l >> 4, m = l & 15;
  // layer 1: [16x96] @ [96x128]
#pragma unroll
  for (int nt = 0; nt < 8; ++nt) {
    float bn = b1[nt * 16 + m];
    v8f acc;
#pragma unroll
    for (int r = 0; r < 8; ++r) acc[r] = bn;
    acc = wmma_f16(a0, bfrag(sW + W1T_OFF, 96, nt * 16 + m, 0, hl), acc);
    acc = wmma_f16(a1, bfrag(sW + W1T_OFF, 96, nt * 16 + m, 1, hl), acc);
    acc = wmma_f16(a2, bfrag(sW + W1T_OFF, 96, nt * 16 + m, 2, hl), acc);
    acc = tanh8(acc);
    store_act(act, acc, nt, l);
  }
  ds_fence();
#pragma unroll
  for (int kc = 0; kc < 4; ++kc) h1f[kc] = afrag(act + m * HID, kc, hl);
  ds_fence();
  // layer 2: [16x128] @ [128x128]
#pragma unroll
  for (int nt = 0; nt < 8; ++nt) {
    float bn = b2[nt * 16 + m];
    v8f acc;
#pragma unroll
    for (int r = 0; r < 8; ++r) acc[r] = bn;
#pragma unroll
    for (int kc = 0; kc < 4; ++kc)
      acc = wmma_f16(h1f[kc], bfrag(sW + W2T_OFF, 128, nt * 16 + m, kc, hl), acc);
    acc = tanh8(acc);
    h2t[nt] = acc;
    store_act(act, acc, nt, l);
  }
  ds_fence();
  // layer 3: [16x128] @ [128x16] (cols 3..15 zero), B frags pre-hoisted
  {
    float bn = (m < 3) ? b3[m] : 0.0f;
    v8f acc;
#pragma unroll
    for (int r = 0; r < 8; ++r) acc[r] = bn;
#pragma unroll
    for (int kc = 0; kc < 4; ++kc)
      acc = wmma_f16(afrag(act + m * HID, kc, hl), w3f[kc], acc);
    store_c(cc, acc, l);
  }
  ds_fence();
  vel[0] = cc[m * 16 + 0];
  vel[1] = cc[m * 16 + 1];
  vel[2] = cc[m * 16 + 2];
}

// ---------------- main kernel ----------------
__global__ __launch_bounds__(WAVES * 32) void velocity_warp_kernel(
    const float* __restrict__ code, const float* __restrict__ pos,
    const float* __restrict__ t1g, const float* __restrict__ t2g,
    const float* __restrict__ W1g, const float* __restrict__ b1,
    const float* __restrict__ b2, const float* __restrict__ b3,
    const _Float16* __restrict__ ws, float* __restrict__ out) {
  extern __shared__ __align__(32) char smem[];
  _Float16* sW = (_Float16*)smem;                      // 61440 B of weights
  _Float16* sActAll = (_Float16*)(smem + W_BYTES);     // per-wave activations
  float* sCAll = (float*)(smem + W_BYTES + SACT_BYTES);

  int tid = threadIdx.x;
  int w = tid >> 5, l = tid & 31;
  int hl = l >> 4, m = l & 15;
  int p = blockIdx.x * PTS_BLK + w * 16 + m;  // lanes 16..31 mirror lanes 0..15

  // block-cooperative copy of f16 weights into LDS (16B chunks)
  {
    const int4* src = (const int4*)ws;
    int4* dst = (int4*)sW;
    for (int i = tid; i < W_BYTES / 16; i += WAVES * 32) dst[i] = src[i];
  }
  __syncthreads();

  _Float16* act = sActAll + w * (16 * HID);
  float* cc = sCAll + w * (16 * 16);

  // hoisted W3 B fragments (constant all kernel; n = m, nt = 0)
  v16h w3f[4];
#pragma unroll
  for (int kc = 0; kc < 4; ++kc)
    w3f[kc] = bfrag(sW + W3T_OFF, 128, m, kc, hl);

  // constant code A-fragments (K = 0..63), loaded once
  v16h a0, a1;
#pragma unroll
  for (int kc = 0; kc < 2; ++kc) {
    const float* cp = code + (size_t)p * CODE_DIM + kc * 32 + hl * 8;
    v16h f;
#pragma unroll
    for (int e = 0; e < 8; ++e) {
      f[e] = (_Float16)cp[e];
      f[e + 8] = (_Float16)cp[e + 16];
    }
    if (kc == 0) a0 = f; else a1 = f;
  }

  // per-point ODE state (mirrored on both lane halves)
  float px = pos[p * 3 + 0], py = pos[p * 3 + 1], pz = pos[p * 3 + 2];
  float tc = t1g[p];
  float off = t1g[p] - t2g[p];
  float D[9] = {1.f, 0.f, 0.f, 0.f, 1.f, 0.f, 0.f, 0.f, 1.f};

  v16h h1f[4];
  v8f h2t[8];
  float vel[3];

#pragma unroll 1
  for (int step = 0; step < N_STEPS; ++step) {
    float dt = copysignf(fminf(fabsf(off), DT_MAX), off);

    // ---- eval A: v0 at (xyz, t) ----
    v16h a2;
#pragma unroll
    for (int e = 0; e < 16; ++e) a2[e] = (_Float16)0.0f;
    a2[0] = (_Float16)((hl == 0) ? px : 0.0f);
    a2[1] = (_Float16)((hl == 0) ? py : 0.0f);
    a2[2] = (_Float16)((hl == 0) ? pz : 0.0f);
    a2[3] = (_Float16)((hl == 0) ? tc : 0.0f);
    mlp_eval(sW, b1, b2, b3, act, cc, l, a0, a1, a2, w3f, h1f, h2t, vel);

    float pmx = px - 0.5f * dt * vel[0];
    float pmy = py - 0.5f * dt * vel[1];
    float pmz = pz - 0.5f * dt * vel[2];
    float tm = tc - 0.5f * dt;

    // ---- eval B: velocity + spatial Jacobian at midpoint ----
#pragma unroll
    for (int e = 0; e < 16; ++e) a2[e] = (_Float16)0.0f;
    a2[0] = (_Float16)((hl == 0) ? pmx : 0.0f);
    a2[1] = (_Float16)((hl == 0) ? pmy : 0.0f);
    a2[2] = (_Float16)((hl == 0) ? pmz : 0.0f);
    a2[3] = (_Float16)((hl == 0) ? tm : 0.0f);
    mlp_eval(sW, b1, b2, b3, act, cc, l, a0, a1, a2, w3f, h1f, h2t, vel);

    // s1 = 1 - h1m^2 in A layout (tanh' factor, layer 1)
    v16h s1[4];
#pragma unroll
    for (int kc = 0; kc < 4; ++kc) {
      v16h h = h1f[kc];
#pragma unroll
      for (int e = 0; e < 16; ++e)
        s1[kc][e] = (_Float16)1.0f - h[e] * h[e];
    }

    // 3 forward-mode JVPs: tangent = e_x/e_y/e_z  =>  g1 = W1_row(64+j) * s1
    float jac[3][3];  // [out o][dir j]
#pragma unroll
    for (int j = 0; j < 3; ++j) {
      v16h g1[4];
#pragma unroll
      for (int kc = 0; kc < 4; ++kc) {
        const float* wr = W1g + (size_t)(CODE_DIM + j) * HID + kc * 32 + hl * 8;
        v16h wf;
#pragma unroll
        for (int e = 0; e < 8; ++e) {
          wf[e] = (_Float16)wr[e];
          wf[e + 8] = (_Float16)wr[e + 16];
        }
        g1[kc] = wf * s1[kc];
      }
      // g2 = (g1 @ W2) * (1 - h2m^2), stored to LDS in f16
#pragma unroll
      for (int nt = 0; nt < 8; ++nt) {
        v8f acc;
#pragma unroll
        for (int r = 0; r < 8; ++r) acc[r] = 0.0f;
#pragma unroll
        for (int kc = 0; kc < 4; ++kc)
          acc = wmma_f16(g1[kc], bfrag(sW + W2T_OFF, 128, nt * 16 + m, kc, hl),
                         acc);
#pragma unroll
        for (int r = 0; r < 8; ++r)
          acc[r] *= (1.0f - h2t[nt][r] * h2t[nt][r]);
        store_act(act, acc, nt, l);
      }
      ds_fence();
      // jac column j = g2 @ W3
      v8f jacc;
#pragma unroll
      for (int r = 0; r < 8; ++r) jacc[r] = 0.0f;
#pragma unroll
      for (int kc = 0; kc < 4; ++kc)
        jacc = wmma_f16(afrag(act + m * HID, kc, hl), w3f[kc], jacc);
      store_c(cc, jacc, l);
      ds_fence();
      jac[0][j] = cc[m * 16 + 0];
      jac[1][j] = cc[m * 16 + 1];
      jac[2][j] = cc[m * 16 + 2];
    }

    // deform -= dt * (jac3x3 @ deform);  state integration
    float nD[9];
#pragma unroll
    for (int o = 0; o < 3; ++o)
#pragma unroll
      for (int i = 0; i < 3; ++i)
        nD[o * 3 + i] =
            D[o * 3 + i] - dt * (jac[o][0] * D[0 * 3 + i] +
                                 jac[o][1] * D[1 * 3 + i] +
                                 jac[o][2] * D[2 * 3 + i]);
#pragma unroll
    for (int k = 0; k < 9; ++k) D[k] = nD[k];

    px -= dt * vel[0];
    py -= dt * vel[1];
    pz -= dt * vel[2];
    tc -= dt;
    off -= dt;
  }

  if (hl == 0) {
    out[p * 3 + 0] = px;
    out[p * 3 + 1] = py;
    out[p * 3 + 2] = pz;
    float* od = out + (size_t)NPTS * 3 + (size_t)p * 9;
#pragma unroll
    for (int k = 0; k < 9; ++k) od[k] = D[k];
  }
}

// ---------------- launch ----------------
extern "C" void kernel_launch(void* const* d_in, const int* in_sizes, int n_in,
                              void* d_out, int out_size, void* d_ws,
                              size_t ws_size, hipStream_t stream) {
  const float* code = (const float*)d_in[0];
  const float* pos = (const float*)d_in[1];
  const float* t1 = (const float*)d_in[2];
  const float* t2 = (const float*)d_in[3];
  const float* W1 = (const float*)d_in[4];
  const float* b1 = (const float*)d_in[5];
  const float* W2 = (const float*)d_in[6];
  const float* b2 = (const float*)d_in[7];
  const float* W3 = (const float*)d_in[8];
  const float* b3 = (const float*)d_in[9];
  _Float16* ws = (_Float16*)d_ws;

  prep_weights<<<64, 256, 0, stream>>>(W1, W2, W3, ws);

  int nblk = NPTS / PTS_BLK;
  velocity_warp_kernel<<<nblk, WAVES * 32, SMEM_BYTES, stream>>>(
      code, pos, t1, t2, W1, b1, b2, b3, ws, (float*)d_out);
}
